// MaskedAttention_62113817035141
// MI455X (gfx1250) — matrix operand order repaired and checked by reading, not compiled, not run on hardware
//
#include <hip/hip_runtime.h>
#include <hip/hip_bf16.h>

// MI455X (gfx1250) fused causal attention.
// Pass A: LDS tile-transpose weights to f16 [n][k] (Wq pre-scaled by 1/sqrt(D)).
// Pass B: QKV projection via v_wmma_f32_16x16x32_f16; B-fragments preloaded so
//         WMMAs retire against descending load counts; outputs staged in LDS and
//         written with coalesced global_store_b128 (Q/K row-major, V as [B,D,T]).
// Pass C: flash attention, one wave per 16-query tile, 32-key blocks, online
//         softmax; K/V fragments preloaded, V loads hidden under softmax VALU.
//         Query tiles are pair-remapped (t/2 <-> 127-t/2) so each SIMD's waves
//         carry a balanced share of the causal triangle.

typedef __attribute__((ext_vector_type(16))) _Float16 v16h;
typedef __attribute__((ext_vector_type(8)))  _Float16 v8h;
typedef __attribute__((ext_vector_type(8)))  float    v8f;

#define EMBED 2048   // C
#define HD    128    // head dim D
#define NB    8      // batch
#define SEQ   2048   // T
#define MTILES (NB * SEQ / 16)   // 1024 row tiles of 16

static __device__ __forceinline__ v8f wmma_f16(v16h a, v16h b, v8f c) {
  // 8 args: (neg_a, A, neg_b, B, c_mod, C, reuse_a, reuse_b)
  return __builtin_amdgcn_wmma_f32_16x16x32_f16(false, a, false, b, (short)0, c,
                                                false, false);
}

// ---------------------------------------------------------------------------
// Pass A: W[k][n] (f32) -> Wt[mat][n][k] (f16), coalesced both sides via LDS.
// ---------------------------------------------------------------------------
__global__ __launch_bounds__(256) void wconv_kernel(const float* __restrict__ Wq,
                                                    const float* __restrict__ Wk,
                                                    const float* __restrict__ Wv,
                                                    _Float16* __restrict__ Wt) {
  __shared__ float tile[32][33];   // +1 pad: no bank conflicts on transpose read
  int mat = blockIdx.z;
  int k0  = blockIdx.x * 32;       // EMBED/32 blocks
  int n0  = blockIdx.y * 32;       // HD/32 blocks
  int tx  = threadIdx.x;           // 0..31
  int ty  = threadIdx.y;           // 0..7
  const float* W = (mat == 0) ? Wq : (mat == 1) ? Wk : Wv;
  float s = (mat == 0) ? 0.08838834764831845f : 1.0f;  // 1/sqrt(128)

#pragma unroll
  for (int i = 0; i < 4; ++i) {
    int r = ty + i * 8;
    tile[r][tx] = W[(size_t)(k0 + r) * HD + (n0 + tx)];   // coalesced in tx
  }
  __syncthreads();
#pragma unroll
  for (int i = 0; i < 4; ++i) {
    int r = ty + i * 8;            // n index within tile
    Wt[(size_t)mat * HD * EMBED + (size_t)(n0 + r) * EMBED + k0 + tx] =
        (_Float16)(tile[tx][r] * s);                      // coalesced in tx
  }
}

// ---------------------------------------------------------------------------
// Pass B: QKV projection. One wave computes a 16x128 strip of one of Q/K/V.
// ---------------------------------------------------------------------------
__global__ __launch_bounds__(256) void qkv_proj_kernel(const float* __restrict__ x,
                                                       const _Float16* __restrict__ Wt,
                                                       _Float16* __restrict__ Qh,
                                                       _Float16* __restrict__ Kh,
                                                       _Float16* __restrict__ VhT) {
  __shared__ __align__(16) _Float16 sbuf[8][2048];  // 4KB/wave store-staging

  int wave = (int)((blockIdx.x * blockDim.x + threadIdx.x) >> 5);
  int lane = threadIdx.x & 31;
  int wv   = threadIdx.x >> 5;
  int mat  = wave / MTILES;           // 0=Q, 1=K, 2=V
  int mt   = wave % MTILES;
  int b    = mt >> 7;                 // SEQ/16 = 128 tiles per batch
  int qt   = mt & 127;
  int rowbase = b * SEQ + qt * 16;

  const _Float16* Wm = Wt + (size_t)mat * HD * EMBED;

  int m    = lane & 15;               // A row within tile / B column within tile
  int hi   = lane >> 4;               // lane group
  int koff = hi * 16;                 // A-fragment K sub-offset
  int bk   = hi * 16;                 // B-fragment K sub-offset

  const float* xr = x + (size_t)(rowbase + m) * EMBED;

  v8f acc[8] = {};
#pragma unroll 1
  for (int kb = 0; kb < EMBED; kb += 32) {
    // Preload all 8 B-fragments (one load clause, WMMAs overlap the tail).
    v16h bb[8];
#pragma unroll
    for (int nt = 0; nt < 8; ++nt) {
      int n = nt * 16 + m;
      bb[nt] = *(const v16h*)(Wm + (size_t)n * EMBED + kb + bk);
    }
    // A fragment: x tile 16x32, f32 -> f16. Two contiguous 8-float runs per lane.
    v16h a;
    const float* p0 = xr + kb + koff;
    const float* p1 = xr + kb + 8 + koff;
#pragma unroll
    for (int i = 0; i < 8; ++i) {
      a[i]     = (_Float16)p0[i];
      a[i + 8] = (_Float16)p1[i];
    }
#pragma unroll
    for (int nt = 0; nt < 8; ++nt)
      acc[nt] = wmma_f16(a, bb[nt], acc[nt]);
  }

  // Stage tile in LDS (C/D layout scatter), then coalesced b128 global stores.
  if (mat != 2) {
    // row-major [16][128]
#pragma unroll
    for (int nt = 0; nt < 8; ++nt)
#pragma unroll
      for (int r = 0; r < 8; ++r) {
        int M = hi ? (r + 8) : r;
        sbuf[wv][M * HD + nt * 16 + m] = (_Float16)acc[nt][r];
      }
    asm volatile("s_wait_dscnt 0x0" ::: "memory");
    _Float16* dst = ((mat == 0) ? Qh : Kh) + (size_t)rowbase * HD;
#pragma unroll
    for (int i = 0; i < 8; ++i) {
      int o = (i * 32 + lane) * 8;    // 2048 halves = contiguous 4KB tile
      *(v8h*)(dst + o) = *(const v8h*)&sbuf[wv][o];
    }
  } else {
    // transposed [128][16] -> VhT[B][D][T]
#pragma unroll
    for (int nt = 0; nt < 8; ++nt)
#pragma unroll
      for (int r = 0; r < 8; ++r) {
        int M = hi ? (r + 8) : r;
        sbuf[wv][(nt * 16 + m) * 16 + M] = (_Float16)acc[nt][r];
      }
    asm volatile("s_wait_dscnt 0x0" ::: "memory");
#pragma unroll
    for (int i = 0; i < 8; ++i) {
      int d   = i * 16 + (lane >> 1);
      int off = (lane & 1) * 8;
      *(v8h*)(VhT + ((size_t)b * HD + d) * SEQ + qt * 16 + off) =
          *(const v8h*)&sbuf[wv][d * 16 + off];
    }
  }
}

// ---------------------------------------------------------------------------
// Pass C: flash attention. One wave per (batch, 16-query tile), blocks of 32 keys.
// ---------------------------------------------------------------------------
__global__ __launch_bounds__(256) void fattn_kernel(const _Float16* __restrict__ Qh,
                                                    const _Float16* __restrict__ Kh,
                                                    const _Float16* __restrict__ VhT,
                                                    float* __restrict__ out) {
  // 8KB/wave, dual-use: first 1KB = P transpose bounce (f16 [16][32]) during the
  // key loop; whole buffer = f32 [16][128] output staging in the epilogue.
  __shared__ __align__(16) float smem[8][2048];     // 64KB

  int wv   = threadIdx.x >> 5;
  int lane = threadIdx.x & 31;
  int wave = blockIdx.x * 8 + wv;
  int b = wave >> 7;
  int t = wave & 127;
  // Causal-triangle balancing: pair short tiles with long ones so adjacent
  // waves (sharing a SIMD) sum to ~constant work.
  int qt = (t & 1) ? (127 - (t >> 1)) : (t >> 1);
  size_t bT = (size_t)b * SEQ;

  int m  = lane & 15;
  int hi = lane >> 4;

  _Float16* pb = (_Float16*)&smem[wv][0];           // P bounce: [16][32] halves

  // Preload Q A-fragments (16x128, 4 k-steps of 32).
  v16h aQ[4];
#pragma unroll
  for (int ds = 0; ds < 4; ++ds) {
    const _Float16* qp = Qh + (bT + qt * 16 + m) * HD + ds * 32 + hi * 16;
    v8h lo = *(const v8h*)qp;
    v8h hh = *(const v8h*)(qp + 8);
    v16h a;
#pragma unroll
    for (int i = 0; i < 8; ++i) { a[i] = lo[i]; a[i + 8] = hh[i]; }
    aQ[ds] = a;
  }

  v8f acc[8] = {};
  float mrun[8], lrun[8];
#pragma unroll
  for (int r = 0; r < 8; ++r) { mrun[r] = -1e30f; lrun[r] = 0.0f; }

  int nblk = (qt * 16 + 15) / 32 + 1;  // causal limit in 32-key blocks

#pragma unroll 1
  for (int kb2 = 0; kb2 < nblk; ++kb2) {
    int s0 = kb2 * 32;

    // --- preload K B-fragments for keys [s0, s0+32), then 8 score WMMAs ---
    v16h kf[8];
#pragma unroll
    for (int ds = 0; ds < 4; ++ds) {
      kf[2 * ds]     = *(const v16h*)(Kh + (bT + s0 + m)      * HD + ds * 32 + hi * 16);
      kf[2 * ds + 1] = *(const v16h*)(Kh + (bT + s0 + 16 + m) * HD + ds * 32 + hi * 16);
    }
    v8f sc0 = {}, sc1 = {};
#pragma unroll
    for (int ds = 0; ds < 4; ++ds) {
      sc0 = wmma_f16(aQ[ds], kf[2 * ds],     sc0);
      sc1 = wmma_f16(aQ[ds], kf[2 * ds + 1], sc1);
    }

    // --- issue V B-fragment loads now; latency hides under softmax VALU ---
    v16h bv[8];
#pragma unroll
    for (int nt = 0; nt < 8; ++nt) {
      int d = nt * 16 + m;
      bv[nt] = *(const v16h*)(VhT + ((size_t)b * HD + d) * SEQ + s0 + hi * 16);
    }
    // Prefetch next key block's K rows while we do the softmax.
    if (kb2 + 1 < nblk)
      __builtin_prefetch(Kh + (bT + s0 + 32 + m) * HD, 0, 0);

    // --- causal mask + online softmax (rows live in 16-lane groups) ---
    float fac[8];
#pragma unroll
    for (int r = 0; r < 8; ++r) {
      int M  = hi ? (r + 8) : r;
      int qg = qt * 16 + M;
      float v0 = sc0[r];
      float v1 = sc1[r];
      if (s0 + m > qg)      v0 = -1e30f;
      if (s0 + 16 + m > qg) v1 = -1e30f;

      float bm = fmaxf(v0, v1);
#pragma unroll
      for (int off = 1; off < 16; off <<= 1)
        bm = fmaxf(bm, __shfl_xor(bm, off, 32));

      float mnew = fmaxf(mrun[r], bm);
      fac[r] = __expf(mrun[r] - mnew);
      v0 = __expf(v0 - mnew);
      v1 = __expf(v1 - mnew);

      float rs = v0 + v1;
#pragma unroll
      for (int off = 1; off < 16; off <<= 1)
        rs += __shfl_xor(rs, off, 32);

      lrun[r] = lrun[r] * fac[r] + rs;
      mrun[r] = mnew;

      // C-layout -> LDS (row M, cols m and 16+m), re-read below in A-layout.
      pb[M * 32 + m]      = (_Float16)v0;
      pb[M * 32 + 16 + m] = (_Float16)v1;
    }

    // rescale running accumulator
#pragma unroll
    for (int nt = 0; nt < 8; ++nt)
#pragma unroll
      for (int r = 0; r < 8; ++r)
        acc[nt][r] *= fac[r];

    // DS ops are in-order within a wave; wait for the P stores to land.
    asm volatile("s_wait_dscnt 0x0" ::: "memory");

    // Re-read P as a 16x32 f16 A-fragment.
    v16h aP;
    {
      v8h lo = *(const v8h*)&pb[m * 32 + hi * 16];
      v8h hh = *(const v8h*)&pb[m * 32 + hi * 16 + 8];
#pragma unroll
      for (int i = 0; i < 8; ++i) { aP[i] = lo[i]; aP[i + 8] = hh[i]; }
    }

    // --- O += P . V ---
#pragma unroll
    for (int nt = 0; nt < 8; ++nt)
      acc[nt] = wmma_f16(aP, bv[nt], acc[nt]);
  }

  // epilogue: O / l, staged in LDS, then coalesced b128 stores of the 8KB tile.
  float inv[8];
#pragma unroll
  for (int r = 0; r < 8; ++r) inv[r] = 1.0f / lrun[r];
#pragma unroll
  for (int nt = 0; nt < 8; ++nt)
#pragma unroll
    for (int r = 0; r < 8; ++r) {
      int M = hi ? (r + 8) : r;
      smem[wv][M * HD + nt * 16 + m] = acc[nt][r] * inv[r];
    }
  asm volatile("s_wait_dscnt 0x0" ::: "memory");
  float* dst = out + (bT + qt * 16) * HD;
#pragma unroll
  for (int i = 0; i < 16; ++i) {
    int o = (i * 32 + lane) * 4;      // 2048 floats = contiguous 8KB tile
    *(float4*)(dst + o) = *(const float4*)&smem[wv][o];
  }
}

// ---------------------------------------------------------------------------
extern "C" void kernel_launch(void* const* d_in, const int* in_sizes, int n_in,
                              void* d_out, int out_size, void* d_ws, size_t ws_size,
                              hipStream_t stream) {
  (void)in_sizes; (void)n_in; (void)out_size; (void)ws_size;
  const float* x  = (const float*)d_in[0];
  const float* Wq = (const float*)d_in[1];
  const float* Wk = (const float*)d_in[2];
  const float* Wv = (const float*)d_in[3];
  float* out = (float*)d_out;

  // Workspace layout (halves): Qh | Kh | VhT | Wt  -> ~13.5 MB total
  size_t NTD = (size_t)NB * SEQ * HD;          // 2,097,152 elements each
  _Float16* Qh  = (_Float16*)d_ws;
  _Float16* Kh  = Qh + NTD;
  _Float16* VhT = Kh + NTD;
  _Float16* Wt  = VhT + NTD;                   // 3 * HD * EMBED halves

  wconv_kernel<<<dim3(EMBED / 32, HD / 32, 3), dim3(32, 8), 0, stream>>>(Wq, Wk, Wv, Wt);
  qkv_proj_kernel<<<(3 * MTILES) / 8, 256, 0, stream>>>(x, Wt, Qh, Kh, VhT);
  fattn_kernel<<<MTILES / 8, 256, 0, stream>>>(Qh, Kh, VhT, out);
}
